// MiMoV2MoE_37881611550759
// MI455X (gfx1250) — matrix operand order repaired and checked by reading, not compiled, not run on hardware
//
#include <hip/hip_runtime.h>
#include <hip/hip_bf16.h>
#include <stdint.h>

// Problem constants (match reference)
#define T_TOK 16384
#define HDIM  1024
#define NEXP  64
#define IDIM  768
#define TOPK  8
#define NGRP  8
#define KGRP  4

#define MBLK  64            // tokens per expert workgroup
#define XT_S  (HDIM + 8)    // padded LDS stride (bank-conflict-free b128 frag loads)
#define HT_S  (IDIM + 8)

#if defined(__has_builtin)
#if __has_builtin(__builtin_amdgcn_global_load_async_to_lds_b128) && \
    __has_builtin(__builtin_amdgcn_s_wait_asynccnt)
#define HAVE_ASYNC_LDS 1
#endif
#endif

typedef __attribute__((ext_vector_type(16))) __bf16 v16bf;
typedef __attribute__((ext_vector_type(8)))  float  v8f;
typedef __attribute__((ext_vector_type(4)))  int    v4i;

#define AS_GLOBAL __attribute__((address_space(1)))
#define AS_LOCAL  __attribute__((address_space(3)))

// ---------- helpers ----------
__device__ __forceinline__ float fastrcp(float x) {
#if __has_builtin(__builtin_amdgcn_rcpf)
  return __builtin_amdgcn_rcpf(x);      // v_rcp_f32
#else
  return 1.f / x;
#endif
}

// float -> bf16 bits, round-to-nearest-even
__device__ __forceinline__ uint16_t f2bfu(float f) {
  union { float f; uint32_t u; } v; v.f = f;
  uint32_t r = v.u + 0x7FFFu + ((v.u >> 16) & 1u);
  return (uint16_t)(r >> 16);
}
__device__ __forceinline__ __bf16 f2bf(float f) {
  uint16_t h = f2bfu(f);
  __bf16 o; __builtin_memcpy(&o, &h, 2);
  return o;
}

// A-fragment (16x32 bf16, M = lane%16) from row-major bf16 LDS row.
// Layout: lanes<16 hold K = 0..15; lanes>=16 hold K = 16..31 (matches B fragment).
__device__ __forceinline__ v16bf ldsFragA(const __bf16* p, int half) {
  v16bf f;
  __builtin_memcpy((char*)&f, p + half * 16, 32);            // 2x ds_load_b128
  return f;
}

// B-fragment (32x16, N = lane%16) from row-major [N][K] *bf16* global row:
// lanes<16 hold K=0..15, lanes>=16 hold K=16..31 -> one contiguous 32B run.
__device__ __forceinline__ v16bf gblFragBbf(const __bf16* p, int half) {
  v16bf f;
  __builtin_memcpy(&f, p + half * 16, 32);                   // 2x global_load_b128
  return f;
}

// Same fragment from *fp32* global row (fallback path, converts on the fly).
__device__ __forceinline__ v16bf gblFragBf32(const float* p, int half) {
  const float4* q = (const float4*)(p + half * 16);
  float4 a = q[0], b = q[1], c = q[2], d = q[3];
  v16bf f;
  f[0]  = f2bf(a.x); f[1]  = f2bf(a.y); f[2]  = f2bf(a.z); f[3]  = f2bf(a.w);
  f[4]  = f2bf(b.x); f[5]  = f2bf(b.y); f[6]  = f2bf(b.z); f[7]  = f2bf(b.w);
  f[8]  = f2bf(c.x); f[9]  = f2bf(c.y); f[10] = f2bf(c.z); f[11] = f2bf(c.w);
  f[12] = f2bf(d.x); f[13] = f2bf(d.y); f[14] = f2bf(d.z); f[15] = f2bf(d.w);
  return f;
}

// ---------- utility kernels ----------
__global__ void zero_f32(float* p, long long n) {
  long long i = (long long)blockIdx.x * blockDim.x + threadIdx.x;
  if (i < n) p[i] = 0.f;
}
__global__ void zero_i32(int* p, int n) {
  int i = blockIdx.x * blockDim.x + threadIdx.x;
  if (i < n) p[i] = 0;
}

// One-shot fp32 -> bf16 tensor convert (vectorized: float4 in, ushort4 out)
__global__ void cvt_bf16(const float* __restrict__ src, __bf16* __restrict__ dst,
                         long long n4) {
  long long i = (long long)blockIdx.x * blockDim.x + threadIdx.x;
  if (i < n4) {
    float4 v = ((const float4*)src)[i];
    ushort4 o = make_ushort4(f2bfu(v.x), f2bfu(v.y), f2bfu(v.z), f2bfu(v.w));
    ((ushort4*)dst)[i] = o;                                  // global_store_b64
  }
}

// ---------------- Gate + grouped top-k + expert scatter -----------------
__global__ __launch_bounds__(64)
void gate_kernel(const float* __restrict__ x, const float* __restrict__ gw,
                 const float* __restrict__ bias, int* __restrict__ cnt,
                 int* __restrict__ tokList, float* __restrict__ wList) {
  const int t   = blockIdx.x;
  const int tid = threadIdx.x;            // 64 threads = 2 waves
  __shared__ float xs[HDIM];
  __shared__ float sc[NEXP];              // sigmoid scores
  __shared__ float sb[NEXP];              // biased scores
  __shared__ int   selE[TOPK];
  __shared__ float selW[TOPK];

  for (int h = tid; h < HDIM; h += 64) xs[h] = x[(size_t)t * HDIM + h];
  __syncthreads();

  const float* wrow = gw + (size_t)tid * HDIM;
  float acc = 0.f;
  #pragma unroll 4
  for (int h = 0; h < HDIM; h += 4)
    acc += xs[h] * wrow[h] + xs[h + 1] * wrow[h + 1] +
           xs[h + 2] * wrow[h + 2] + xs[h + 3] * wrow[h + 3];
  float s = fastrcp(1.f + __expf(-acc));
  sc[tid] = s;
  sb[tid] = s + bias[tid];
  __syncthreads();

  if (tid == 0) {
    float gs[NGRP];
    for (int g = 0; g < NGRP; ++g) {
      float m1 = -1e30f, m2 = -1e30f;
      for (int j = 0; j < NEXP / NGRP; ++j) {
        float v = sb[g * (NEXP / NGRP) + j];
        if (v > m1) { m2 = m1; m1 = v; } else if (v > m2) m2 = v;
      }
      gs[g] = m1 + m2;
    }
    unsigned gmask = 0;
    for (int it = 0; it < KGRP; ++it) {
      float b = -1e30f; int bi = 0;
      for (int g = 0; g < NGRP; ++g)
        if (!((gmask >> g) & 1u) && gs[g] > b) { b = gs[g]; bi = g; }
      gmask |= 1u << bi;
    }
    unsigned long long taken = 0; float wsum = 0.f;
    for (int it = 0; it < TOPK; ++it) {
      float b = -1e30f; int bi = 0;
      for (int e = 0; e < NEXP; ++e) {
        if (!((gmask >> (e >> 3)) & 1u)) continue;
        if ((taken >> e) & 1ull) continue;
        if (sb[e] > b) { b = sb[e]; bi = e; }
      }
      taken |= 1ull << bi;
      selE[it] = bi; selW[it] = sc[bi]; wsum += sc[bi];
    }
    float inv = fastrcp(wsum + 1e-20f);
    for (int it = 0; it < TOPK; ++it) selW[it] *= inv;
  }
  __syncthreads();

  if (tid < TOPK) {
    int e   = selE[tid];
    int pos = atomicAdd(&cnt[e], 1);
    tokList[(size_t)e * T_TOK + pos] = t;
    wList [(size_t)e * T_TOK + pos] = selW[tid];
  }
}

// --------- Fused expert FFN: h = silu(X Wg^T) * (X Wu^T); out += c * h Wd^T ---------
// PRE=true : bf16-preconverted operands in workspace (pure b128 traffic in k-loop)
// PRE=false: fp32 operands, convert to bf16 on the fly
template <bool PRE>
__global__ __launch_bounds__(256)
void expert_kernel(const float*  __restrict__ x,   const __bf16* __restrict__ xbf,
                   const float*  __restrict__ wg,  const float*  __restrict__ wu,
                   const float*  __restrict__ wd,
                   const __bf16* __restrict__ wgb, const __bf16* __restrict__ wub,
                   const __bf16* __restrict__ wdb,
                   const int*    __restrict__ cnt,
                   const int*    __restrict__ tokList,
                   const float*  __restrict__ wList,
                   float*        __restrict__ out) {
  const int e    = blockIdx.y;
  const int blk  = blockIdx.x;
  const int nTok = cnt[e];
  if (blk * MBLK >= nTok) return;

  extern __shared__ char smem[];
  __bf16* xt  = (__bf16*)smem;                                        // [MBLK][XT_S]
  __bf16* ht  = (__bf16*)(smem + (size_t)MBLK * XT_S * 2);            // [MBLK][HT_S]
  int*    tok = (int*)  (smem + (size_t)MBLK * (XT_S + HT_S) * 2);    // [MBLK]
  float*  twt = (float*)(tok + MBLK);                                 // [MBLK]

  const int tid  = threadIdx.x;
  const int lane = tid & 31;
  const int wave = tid >> 5;     // 8 waves
  const int half = lane >> 4;
  const int lr   = lane & 15;
  const int mi   = wave >> 1;    // 0..3  (16-row M slab)
  const int nj   = wave & 1;     // 0..1  (N half)

  if (tid < MBLK) {
    int gi = blk * MBLK + tid;
    if (gi < nTok) {
      tok[tid] = tokList[(size_t)e * T_TOK + gi];
      twt[tid] = wList [(size_t)e * T_TOK + gi];
    } else { tok[tid] = -1; twt[tid] = 0.f; }
  }
  __syncthreads();

  // Gather X tile -> bf16 LDS
  if constexpr (PRE) {
#if defined(HAVE_ASYNC_LDS)
    // Async DMA copy global->LDS (ASYNCcnt path), 16B per lane per iter.
    for (int idx = tid; idx < MBLK * (HDIM / 8); idx += 256) {
      int r  = idx >> 7;                 // HDIM/8 = 128 chunks per row
      int c8 = (idx & 127) * 8;
      int tt = tok[r];
      __bf16* dst = xt + (size_t)r * XT_S + c8;
      if (tt >= 0) {
        __builtin_amdgcn_global_load_async_to_lds_b128(
            (AS_GLOBAL v4i*)(xbf + (size_t)tt * HDIM + c8),
            (AS_LOCAL  v4i*)dst,
            0, 0);
      } else {
        uint32_t z[4] = {0, 0, 0, 0};
        __builtin_memcpy(dst, z, 16);
      }
    }
    __builtin_amdgcn_s_wait_asynccnt(0);
#else
    for (int idx = tid; idx < MBLK * (HDIM / 8); idx += 256) {
      int r  = idx >> 7;                 // HDIM/8 = 128 chunks per row
      int c8 = (idx & 127) * 8;
      int tt = tok[r];
      char buf[16] = {0};
      if (tt >= 0) __builtin_memcpy(buf, xbf + (size_t)tt * HDIM + c8, 16);
      __builtin_memcpy(xt + (size_t)r * XT_S + c8, buf, 16);
    }
#endif
  } else {
    for (int idx = tid; idx < MBLK * (HDIM / 4); idx += 256) {
      int r  = idx >> 8;                 // HDIM/4 = 256 chunks per row
      int c4 = (idx & 255) * 4;
      int tt = tok[r];
      float4 v = make_float4(0.f, 0.f, 0.f, 0.f);
      if (tt >= 0) v = *(const float4*)(x + (size_t)tt * HDIM + c4);
      __bf16* d = xt + (size_t)r * XT_S + c4;
      d[0] = f2bf(v.x); d[1] = f2bf(v.y); d[2] = f2bf(v.z); d[3] = f2bf(v.w);
    }
  }
  __syncthreads();

  // ---- GEMM 1+2: h[M=64][I=768] = silu(X Wg^T) * (X Wu^T) ----
  const size_t eGU = (size_t)e * IDIM * HDIM;
  const __bf16* aBase = xt + (size_t)(mi * 16 + lr) * XT_S;
  for (int nt = 0; nt < (IDIM / 2) / 16; ++nt) {     // 24 N-tiles per wave
    const int n0 = nj * (IDIM / 2) + nt * 16;
    const size_t rowOff = eGU + (size_t)(n0 + lr) * HDIM;
    v8f accg = {}; v8f accu = {};
    if constexpr (PRE) {
      const __bf16* gRow = wgb + rowOff;
      const __bf16* uRow = wub + rowOff;
      __builtin_prefetch(gRow + (size_t)16 * HDIM, 0, 1);   // global_prefetch_b8
      __builtin_prefetch(uRow + (size_t)16 * HDIM, 0, 1);
      #pragma unroll 4
      for (int k0 = 0; k0 < HDIM; k0 += 32) {
        v16bf a  = ldsFragA(aBase + k0, half);
        v16bf bg = gblFragBbf(gRow + k0, half);
        v16bf bu = gblFragBbf(uRow + k0, half);
        accg = __builtin_amdgcn_wmma_f32_16x16x32_bf16(false, a, false, bg, (short)0, accg, false, false);
        accu = __builtin_amdgcn_wmma_f32_16x16x32_bf16(false, a, false, bu, (short)0, accu, false, false);
      }
    } else {
      const float* gRow = wg + rowOff;
      const float* uRow = wu + rowOff;
      __builtin_prefetch(gRow + (size_t)16 * HDIM, 0, 1);
      __builtin_prefetch(uRow + (size_t)16 * HDIM, 0, 1);
      #pragma unroll 2
      for (int k0 = 0; k0 < HDIM; k0 += 32) {
        v16bf a  = ldsFragA(aBase + k0, half);
        v16bf bg = gblFragBf32(gRow + k0, half);
        v16bf bu = gblFragBf32(uRow + k0, half);
        accg = __builtin_amdgcn_wmma_f32_16x16x32_bf16(false, a, false, bg, (short)0, accg, false, false);
        accu = __builtin_amdgcn_wmma_f32_16x16x32_bf16(false, a, false, bu, (short)0, accu, false, false);
      }
    }
    #pragma unroll
    for (int v = 0; v < 8; ++v) {                     // C layout: M = v + 8*half, N = lr
      float g   = accg[v];
      float sig = fastrcp(1.f + __expf(-g));          // v_rcp_f32, no IEEE div chain
      float h   = g * sig * accu[v];                  // silu(g) * u
      ht[(size_t)(mi * 16 + v + 8 * half) * HT_S + n0 + lr] = f2bf(h);
    }
  }
  __syncthreads();

  // ---- GEMM 3: out[M=64][H=1024] += c * h Wd^T ----
  const size_t eD = (size_t)e * HDIM * IDIM;
  const __bf16* hBase = ht + (size_t)(mi * 16 + lr) * HT_S;
  for (int nt = 0; nt < (HDIM / 2) / 16; ++nt) {     // 32 N-tiles per wave
    const int n0 = nj * (HDIM / 2) + nt * 16;
    const size_t rowOff = eD + (size_t)(n0 + lr) * IDIM;
    v8f acc = {};
    if constexpr (PRE) {
      const __bf16* dRow = wdb + rowOff;
      __builtin_prefetch(dRow + (size_t)16 * IDIM, 0, 1);
      #pragma unroll 4
      for (int k0 = 0; k0 < IDIM; k0 += 32) {
        v16bf a = ldsFragA(hBase + k0, half);
        v16bf b = gblFragBbf(dRow + k0, half);
        acc = __builtin_amdgcn_wmma_f32_16x16x32_bf16(false, a, false, b, (short)0, acc, false, false);
      }
    } else {
      const float* dRow = wd + rowOff;
      __builtin_prefetch(dRow + (size_t)16 * IDIM, 0, 1);
      #pragma unroll 2
      for (int k0 = 0; k0 < IDIM; k0 += 32) {
        v16bf a = ldsFragA(hBase + k0, half);
        v16bf b = gblFragBf32(dRow + k0, half);
        acc = __builtin_amdgcn_wmma_f32_16x16x32_bf16(false, a, false, b, (short)0, acc, false, false);
      }
    }
    #pragma unroll
    for (int v = 0; v < 8; ++v) {
      int rloc = mi * 16 + v + 8 * half;
      int tt   = tok[rloc];
      if (tt >= 0)
        atomicAdd(out + (size_t)tt * HDIM + n0 + lr, acc[v] * twt[rloc]);
    }
  }
}

extern "C" void kernel_launch(void* const* d_in, const int* in_sizes, int n_in,
                              void* d_out, int out_size, void* d_ws, size_t ws_size,
                              hipStream_t stream) {
  (void)in_sizes; (void)n_in; (void)out_size;
  const float* x    = (const float*)d_in[0];
  const float* gw   = (const float*)d_in[1];
  const float* bias = (const float*)d_in[2];
  const float* wg   = (const float*)d_in[3];
  const float* wu   = (const float*)d_in[4];
  const float* wd   = (const float*)d_in[5];
  float* out = (float*)d_out;

  // ---- workspace layout (all 256B aligned) ----
  const size_t ROUTE_I = (size_t)NEXP * T_TOK;           // ints / floats per list
  const size_t X_ELEMS = (size_t)T_TOK * HDIM;           // 16.8M
  const size_t W_ELEMS = (size_t)NEXP * IDIM * HDIM;     // 50.3M per tensor
  char*  ws      = (char*)d_ws;
  int*   cnt     = (int*)ws;                             // 64 ints (pad 256B)
  int*   tokList = (int*)(ws + 256);
  float* wList   = (float*)(ws + 256 + ROUTE_I * 4);
  size_t offXbf  = 256 + ROUTE_I * 8;                    // 8,388,864
  __bf16* xbf = (__bf16*)(ws + offXbf);
  __bf16* wgb = (__bf16*)(ws + offXbf + X_ELEMS * 2);
  __bf16* wub = (__bf16*)(ws + offXbf + X_ELEMS * 2 + W_ELEMS * 2);
  __bf16* wdb = (__bf16*)(ws + offXbf + X_ELEMS * 2 + W_ELEMS * 4);
  const size_t NEED = offXbf + X_ELEMS * 2 + W_ELEMS * 6;  // ~344 MB
  const bool pre = (ws_size >= NEED);

  // 1) zero output + counters (harness does not re-zero between replays)
  long long nOut = (long long)T_TOK * HDIM;
  zero_f32<<<(unsigned)((nOut + 255) / 256), 256, 0, stream>>>(out, nOut);
  zero_i32<<<1, NEXP, 0, stream>>>(cnt, NEXP);

  // 2) gating + routing scatter
  gate_kernel<<<T_TOK, 64, 0, stream>>>(x, gw, bias, cnt, tokList, wList);

  // 3) optional one-shot bf16 pre-conversion of X and all expert weights
  if (pre) {
    long long x4 = (long long)(X_ELEMS / 4), w4 = (long long)(W_ELEMS / 4);
    cvt_bf16<<<(unsigned)((x4 + 255) / 256), 256, 0, stream>>>(x,  xbf, x4);
    cvt_bf16<<<(unsigned)((w4 + 255) / 256), 256, 0, stream>>>(wg, wgb, w4);
    cvt_bf16<<<(unsigned)((w4 + 255) / 256), 256, 0, stream>>>(wu, wub, w4);
    cvt_bf16<<<(unsigned)((w4 + 255) / 256), 256, 0, stream>>>(wd, wdb, w4);
  }

  // 4) fused expert FFN, expert-major grid (x fastest) for L2 weight reuse
  size_t ldsBytes = (size_t)MBLK * (XT_S + HT_S) * 2 + (size_t)MBLK * 8;
  dim3 grid(T_TOK / MBLK, NEXP);
  if (pre)
    expert_kernel<true><<<grid, 256, ldsBytes, stream>>>(
        x, xbf, wg, wu, wd, wgb, wub, wdb, cnt, tokList, wList, out);
  else
    expert_kernel<false><<<grid, 256, ldsBytes, stream>>>(
        x, xbf, wg, wu, wd, wgb, wub, wdb, cnt, tokList, wList, out);
}
